// Window_MSA_19430432047557
// MI455X (gfx1250) — compile-verified, matrix operand order checked by reading, not run on hardware
//
#include <hip/hip_runtime.h>
#include <hip/hip_bf16.h>
#include <stdint.h>

typedef _Float16 half_t;
typedef __attribute__((ext_vector_type(16))) _Float16 v16h;
typedef __attribute__((ext_vector_type(8)))  float    v8f;

#define WSZ   7
#define NH    12
#define DIMC  384
#define BB    16
#define HH    56
#define WWI   56
#define NTOK  49
#define HDIM  32
#define NWIN  1024                      // 16 batches * 8*8 windows
#define NTOKENS (BB*HH*WWI)             // 50176
#define QSCALE 0.17677669529663687f     // 32^-0.5

// ---------------------------------------------------------------------------
// WMMA helpers
// ---------------------------------------------------------------------------
__device__ __forceinline__ v8f wmma_f16(v16h a, v16h b, v8f c) {
  return __builtin_amdgcn_wmma_f32_16x16x32_f16(false, a, false, b,
                                                (short)0, c, false, false);
}

// 16-bit A/B fragment K index for element e (0..15) in lane `lane` (0..31):
// lanes 0-15 hold K {0..7,16..23}; lanes 16-31 hold K {8..15,24..31}
__device__ __forceinline__ int kmap(int e, int lane) {
  return ((e & 8) << 1) | ((lane & 16) >> 1) | (e & 7);
}

// window index + token-in-window -> flat token index in [B,H,W] order
__device__ __forceinline__ int token_of(int wi, int j) {
  int b  = wi >> 6;
  int wl = wi & 63;
  int wh = wl >> 3, ww = wl & 7;
  int r = j / WSZ, c = j - r * WSZ;
  return (b * HH + wh * WSZ + r) * WWI + ww * WSZ + c;
}

// generic shared pointer -> 32-bit LDS byte offset (LDS aperture is in the
// high 32 bits of the generic address; low 32 bits are the LDS offset)
__device__ __forceinline__ uint32_t lds_off(const void* p) {
  return (uint32_t)(uintptr_t)p;
}

// async global -> LDS copy of 16 bytes per lane (GLOBAL_LOAD_ASYNC_TO_LDS_B128,
// GVS mode: mem = SADDR + VADDR + 0; LDS dest = VDST vgpr). Tracked by ASYNCcnt.
__device__ __forceinline__ void async_copy16(uint32_t lds_dst, uint32_t g_off,
                                             const void* base) {
  asm volatile("global_load_async_to_lds_b128 %0, %1, %2"
               :: "v"(lds_dst), "v"(g_off), "s"(base) : "memory");
}

__device__ __forceinline__ void wait_async0() {
  asm volatile("s_wait_asynccnt 0x0" ::: "memory");
}

// ---------------------------------------------------------------------------
// fp32 -> fp16 conversion (x is read exactly once -> nontemporal loads)
// ---------------------------------------------------------------------------
__global__ __launch_bounds__(256)
void f32_to_f16(const float* __restrict__ in, half_t* __restrict__ out, int n) {
  int i = blockIdx.x * blockDim.x + threadIdx.x;
  int stride = gridDim.x * blockDim.x;
  for (; i < n; i += stride) out[i] = (half_t)__builtin_nontemporal_load(in + i);
}

// ---------------------------------------------------------------------------
// Tiled WMMA GEMM: C[M,N] = A[M,K] * B[K,N] + bias[N]
// block = 256 threads (8 waves), tile 64x64, BK=32, double-buffered LDS with
// async global->LDS staging. wave w: row-tile = w>>1, two n-tiles at (w&1)*2.
// M % 64 == 0, N % 64 == 0, K % 32 == 0 (holds for all uses here).
// ---------------------------------------------------------------------------
template <bool OUT_F16>
__global__ __launch_bounds__(256)
void wmma_gemm(const half_t* __restrict__ A, const half_t* __restrict__ Bm,
               const float* __restrict__ bias, void* __restrict__ Cout,
               int M, int N, int K) {
  __shared__ half_t sA[2][64][40];   // 64x32 tile, padded rows (80B, 16B-aligned)
  __shared__ half_t sB[2][32][72];   // 32x64 tile, padded rows (144B)

  const int t    = threadIdx.x;
  const int lane = t & 31;
  const int wv   = t >> 5;
  const int rt   = wv >> 1;           // 0..3 : 16-row tile
  const int ct0  = (wv & 1) * 2;      // 0 or 2 : first of two 16-col tiles
  const int rowBase = blockIdx.y * 64;
  const int colBase = blockIdx.x * 64;

  v8f acc0 = {}; v8f acc1 = {};

  const int aRow = t >> 2, aSeg = (t & 3) * 8;   // 64 rows x 4 x 8 halves
  const int bRow = t >> 3, bSeg = (t & 7) * 8;   // 32 rows x 8 x 8 halves

  // per-lane LDS destinations (byte addresses) and global byte offsets
  const uint32_t aDst0 = lds_off(&sA[0][aRow][aSeg]);
  const uint32_t bDst0 = lds_off(&sB[0][bRow][bSeg]);
  const uint32_t aBufStride = lds_off(&sA[1][0][0]) - lds_off(&sA[0][0][0]);
  const uint32_t bBufStride = lds_off(&sB[1][0][0]) - lds_off(&sB[0][0][0]);
  const uint32_t aSrc0 = (uint32_t)(((rowBase + aRow) * K + aSeg) * 2);
  const uint32_t bSrc0 = (uint32_t)((bRow * N + colBase + bSeg) * 2);
  const uint32_t aSrcStep = 64u;                 // 32 halves
  const uint32_t bSrcStep = (uint32_t)(32 * N * 2);

  // prologue: stage K-step 0 into buffer 0
  async_copy16(aDst0, aSrc0, A);
  async_copy16(bDst0, bSrc0, Bm);

  const int nsteps = K / 32;
  for (int s = 0; s < nsteps; ++s) {
    const int buf = s & 1;
    wait_async0();        // own async loads into `buf` complete
    __syncthreads();      // all waves' loads visible; buf^1 free to overwrite

    if (s + 1 < nsteps) { // stage next K-step into the other buffer
      async_copy16(aDst0 + (uint32_t)(buf ^ 1) * aBufStride,
                   aSrc0 + (uint32_t)(s + 1) * aSrcStep, A);
      async_copy16(bDst0 + (uint32_t)(buf ^ 1) * bBufStride,
                   bSrc0 + (uint32_t)(s + 1) * bSrcStep, Bm);
    }

    const int mrow = rt * 16 + (lane & 15);
    const int n0   = ct0 * 16 + (lane & 15);
    v16h af, bf0, bf1;
#pragma unroll
    for (int e = 0; e < 16; ++e) {
      const int k = kmap(e, lane);
      af[e]  = sA[buf][mrow][k];
      bf0[e] = sB[buf][k][n0];
      bf1[e] = sB[buf][k][n0 + 16];
    }
    acc0 = wmma_f16(af, bf0, acc0);
    acc1 = wmma_f16(af, bf1, acc1);
    // no trailing barrier: next iteration's wait+barrier precedes any
    // overwrite of the buffer read here
  }

  const int n0 = colBase + ct0 * 16 + (lane & 15);
  const int n1 = n0 + 16;
  const float b0 = bias[n0];
  const float b1 = bias[n1];
  const int mloc = rt * 16 + ((lane >> 4) << 3);
#pragma unroll
  for (int r = 0; r < 8; ++r) {
    const int m = rowBase + mloc + r;
    const float v0 = acc0[r] + b0;
    const float v1 = acc1[r] + b1;
    if (OUT_F16) {
      half_t* C = (half_t*)Cout;
      C[(size_t)m * N + n0] = (half_t)v0;
      C[(size_t)m * N + n1] = (half_t)v1;
    } else {
      float* C = (float*)Cout;
      C[(size_t)m * N + n0] = v0;
      C[(size_t)m * N + n1] = v1;
    }
  }
}

// ---------------------------------------------------------------------------
// Fused window attention: one block per (head, window), 128 threads = 4 waves.
// 49 tokens padded to 64. HD=32 == WMMA K -> single K-step for q*k^T.
// softmax with q-scale folded in; relative-position bias added AFTER softmax
// (faithful to reference quirk). Output stored f16 in token order.
// ---------------------------------------------------------------------------
__global__ __launch_bounds__(128)
void window_attn(const half_t* __restrict__ qkvh,
                 const float* __restrict__ rel_table,
                 half_t* __restrict__ attnh) {
  __shared__ half_t qs[64][40];
  __shared__ half_t ks[64][40];
  __shared__ half_t vs[64][40];
  __shared__ float  Sf[64][65];
  __shared__ half_t Pp[64][72];

  const int head = blockIdx.x;
  const int wi   = blockIdx.y;
  const int t    = threadIdx.x;
  const int lane = t & 31;
  const int wv   = t >> 5;

  // ---- gather q,k,v rows for this (window, head) into LDS, zero-pad 49..63
  {
    const int row = t >> 1;
    const int h16 = (t & 1) * 16;
    if (row < NTOK) {
      const size_t base = (size_t)token_of(wi, row) * (3 * DIMC) + head * HDIM + h16;
      const uint4* q0 = (const uint4*)(qkvh + base);
      const uint4* k0 = (const uint4*)(qkvh + base + DIMC);
      const uint4* v0 = (const uint4*)(qkvh + base + 2 * DIMC);
      *(uint4*)&qs[row][h16]     = q0[0];
      *(uint4*)&qs[row][h16 + 8] = q0[1];
      *(uint4*)&ks[row][h16]     = k0[0];
      *(uint4*)&ks[row][h16 + 8] = k0[1];
      *(uint4*)&vs[row][h16]     = v0[0];
      *(uint4*)&vs[row][h16 + 8] = v0[1];
    } else {
      const uint4 z = {0u, 0u, 0u, 0u};
      *(uint4*)&qs[row][h16] = z; *(uint4*)&qs[row][h16 + 8] = z;
      *(uint4*)&ks[row][h16] = z; *(uint4*)&ks[row][h16 + 8] = z;
      *(uint4*)&vs[row][h16] = z; *(uint4*)&vs[row][h16 + 8] = z;
    }
  }
  __syncthreads();

  // ---- S = q @ k^T : wave wv owns rows [wv*16, wv*16+16), all 4 n-tiles
  {
    const int mrow = wv * 16 + (lane & 15);
    v16h qf;
#pragma unroll
    for (int e = 0; e < 16; ++e) qf[e] = qs[mrow][kmap(e, lane)];
    const int mbase = wv * 16 + ((lane >> 4) << 3);
#pragma unroll
    for (int nt = 0; nt < 4; ++nt) {
      const int n = nt * 16 + (lane & 15);
      v16h kf;
#pragma unroll
      for (int e = 0; e < 16; ++e) kf[e] = ks[n][kmap(e, lane)];  // B[k][n]=K[n][k]
      v8f s = {};
      s = wmma_f16(qf, kf, s);
#pragma unroll
      for (int r = 0; r < 8; ++r) Sf[mbase + r][n] = s[r];
    }
  }
  __syncthreads();

  // ---- per-row softmax (cols < 49), then add rel-pos bias post-softmax
  if (t < 64) {
    const int m = t;
    float mx = -1e30f;
    for (int n = 0; n < NTOK; ++n) mx = fmaxf(mx, Sf[m][n] * QSCALE);
    float sum = 0.f;
    for (int n = 0; n < NTOK; ++n) {
      const float e = __expf(Sf[m][n] * QSCALE - mx);
      Sf[m][n] = e;
      sum += e;
    }
    const float inv = 1.f / sum;
    const int rm = (m < NTOK) ? (m / WSZ) : 0;
    const int cm = (m < NTOK) ? (m - rm * WSZ) : 0;
    for (int n = 0; n < 64; ++n) {
      float p = 0.f;
      if (n < NTOK) {
        p = Sf[m][n] * inv;
        if (m < NTOK) {
          const int rn = n / WSZ, cn = n - rn * WSZ;
          const int idx = (rm - rn + WSZ - 1) * (2 * WSZ - 1) + (cm - cn + WSZ - 1);
          p += rel_table[idx * NH + head];
        }
      }
      Pp[m][n] = (half_t)p;
    }
  }
  __syncthreads();

  // ---- out = P @ V : K=64 (2 steps), N=32 (2 tiles); store rows < 49
  {
    const int mrow = wv * 16 + (lane & 15);
    v8f o0 = {}, o1 = {};
#pragma unroll
    for (int kk = 0; kk < 2; ++kk) {
      v16h pf, vf0, vf1;
#pragma unroll
      for (int e = 0; e < 16; ++e) {
        const int k = kk * 32 + kmap(e, lane);
        pf[e]  = Pp[mrow][k];
        vf0[e] = vs[k][(lane & 15)];        // B[k][n] = V[k][n]
        vf1[e] = vs[k][16 + (lane & 15)];
      }
      o0 = wmma_f16(pf, vf0, o0);
      o1 = wmma_f16(pf, vf1, o1);
    }
    const int mbase = wv * 16 + ((lane >> 4) << 3);
#pragma unroll
    for (int r = 0; r < 8; ++r) {
      const int m = mbase + r;
      if (m < NTOK) {
        const size_t o = (size_t)token_of(wi, m) * DIMC + head * HDIM;
        attnh[o + (lane & 15)]      = (half_t)o0[r];
        attnh[o + 16 + (lane & 15)] = (half_t)o1[r];
      }
    }
  }
}

// ---------------------------------------------------------------------------
// launch
// ---------------------------------------------------------------------------
extern "C" void kernel_launch(void* const* d_in, const int* in_sizes, int n_in,
                              void* d_out, int out_size, void* d_ws, size_t ws_size,
                              hipStream_t stream) {
  const float* x       = (const float*)d_in[0];
  const float* qkv_w   = (const float*)d_in[1];
  const float* qkv_b   = (const float*)d_in[2];
  const float* proj_w  = (const float*)d_in[3];
  const float* proj_b  = (const float*)d_in[4];
  const float* rel_tab = (const float*)d_in[5];
  (void)in_sizes; (void)n_in; (void)out_size; (void)ws_size;

  char* ws = (char*)d_ws;
  size_t off = 0;
  auto alloc = [&](size_t bytes) -> char* {
    char* p = ws + off;
    off = (off + bytes + 255) & ~(size_t)255;
    return p;
  };
  half_t* xh     = (half_t*)alloc((size_t)NTOKENS * DIMC * sizeof(half_t));
  half_t* wqkvh  = (half_t*)alloc((size_t)DIMC * 3 * DIMC * sizeof(half_t));
  half_t* wprojh = (half_t*)alloc((size_t)DIMC * DIMC * sizeof(half_t));
  half_t* qkvh   = (half_t*)alloc((size_t)NTOKENS * 3 * DIMC * sizeof(half_t));
  half_t* attnh  = (half_t*)alloc((size_t)NTOKENS * DIMC * sizeof(half_t));

  // fp32 -> fp16
  f32_to_f16<<<2048, 256, 0, stream>>>(x, xh, NTOKENS * DIMC);
  f32_to_f16<<<512,  256, 0, stream>>>(qkv_w, wqkvh, DIMC * 3 * DIMC);
  f32_to_f16<<<256,  256, 0, stream>>>(proj_w, wprojh, DIMC * DIMC);

  // QKV projection: [50176,384] x [384,1152] -> f16 qkv
  wmma_gemm<true><<<dim3(3 * DIMC / 64, NTOKENS / 64), 256, 0, stream>>>(
      xh, wqkvh, qkv_b, (void*)qkvh, NTOKENS, 3 * DIMC, DIMC);

  // fused window attention per (head, window)
  window_attn<<<dim3(NH, NWIN), 128, 0, stream>>>(qkvh, rel_tab, attnh);

  // output projection straight into d_out (token order == [B,H,W,C])
  wmma_gemm<false><<<dim3(DIMC / 64, NTOKENS / 64), 256, 0, stream>>>(
      attnh, wprojh, proj_b, d_out, NTOKENS, DIMC, DIMC);
}
